// MHSABlock_2267742732294
// MI455X (gfx1250) — matrix operand order repaired and checked by reading, not compiled, not run on hardware
//
#include <hip/hip_runtime.h>
#include <hip/hip_bf16.h>

// ---------------------------------------------------------------------------
// MHSABlock on gfx1250 (CDNA5, wave32, WMMA + async global->LDS staging).
// ---------------------------------------------------------------------------

typedef _Float16 v16h __attribute__((ext_vector_type(16)));
typedef _Float16 v8h  __attribute__((ext_vector_type(8)));
typedef float    v8f  __attribute__((ext_vector_type(8)));

#define BQ   64      // batch
#define CIN  384
#define HIMG 32
#define DM   768     // model dim
#define HEADS 12
#define HD   64
#define GSZ  16
#define NTOK 257     // 16*16 + 1
#define NP   288     // padded token count (multiple of 32)
#define MLPH 3072
#define KCONV 3456   // 384*9
#define MPIX (BQ*256)    // 16384 conv-output rows
#define MTOK (BQ*NTOK)   // 16448 token rows

// LDS byte offset of a pointer to __shared__ (LDS aperture is in addr[63:32],
// low 32 bits are the LDS offset per CDNA5 ISA flat addressing).
#define LDSOFF(p) ((unsigned)(unsigned long long)(p))

__device__ __forceinline__ void async_g2l_b128(unsigned ldsoff, const _Float16* g) {
    asm volatile("global_load_async_to_lds_b128 %0, %1, off"
                 :: "v"(ldsoff), "v"((unsigned long long)(size_t)g)
                 : "memory");
}

__device__ __forceinline__ void wait_async0() {
#if __has_builtin(__builtin_amdgcn_s_wait_asynccnt)
    __builtin_amdgcn_s_wait_asynccnt(0);
#else
    asm volatile("s_wait_asynccnt 0x0" ::: "memory");
#endif
}

// ---------------- WMMA fragment loaders (16x16x32 f16) ----------------------
// A: 16x32 (MxK) row-major, lane L -> row L&15, K = b..b+15 contiguous, b=(L<16)?0:16
__device__ __forceinline__ v16h load_a_frag(const _Float16* __restrict__ A, long lda,
                                            int m0, int k0, int lane) {
    int row = m0 + (lane & 15);
    int kb  = k0 + ((lane >> 4) << 4);
    const _Float16* p = A + (long)row * lda + kb;
    v8h lo = *(const v8h*)(p);
    v8h hi = *(const v8h*)(p + 8);
    v16h r;
#pragma unroll
    for (int i = 0; i < 8; ++i) { r[i] = lo[i]; r[i + 8] = hi[i]; }
    return r;
}

// B: 32x16 (KxN) sourced from W[N][K] row-major (B[k][n] = W[n][k]).
// lane L -> col L&15, K = base..base+15 contiguous, base=(L<16)?0:16
__device__ __forceinline__ v16h load_b_frag(const _Float16* __restrict__ W, long ldw,
                                            int n0, int k0, int lane) {
    int col = n0 + (lane & 15);
    int kb  = k0 + ((lane >> 4) << 4);
    return *(const v16h*)(W + (long)col * ldw + kb);
}

__device__ __forceinline__ float gelu_exact(float x) {
    return 0.5f * x * (1.0f + erff(x * 0.70710678118654752f));
}

__device__ __forceinline__ float block_sum(float v) {
    __shared__ float sred[8];
#pragma unroll
    for (int off = 16; off > 0; off >>= 1) v += __shfl_xor(v, off, 32);
    int wid = threadIdx.x >> 5;
    __syncthreads();
    if ((threadIdx.x & 31) == 0) sred[wid] = v;
    __syncthreads();
    float t = 0.f;
    int nw = blockDim.x >> 5;
    for (int i = 0; i < nw; ++i) t += sred[i];
    return t;
}

// ---------------- generic utility kernels -----------------------------------
__global__ void cvt_f32_f16(const float* __restrict__ src, _Float16* __restrict__ dst, int n) {
    int i = blockIdx.x * 256 + threadIdx.x;
    if (i < n) dst[i] = (_Float16)src[i];
}

// im2col for conv k3 s2 p1:  x[B,384,32,32] -> A[B*256][3456], col = ci*9+kh*3+kw
__global__ void im2col_f16(const float* __restrict__ x, _Float16* __restrict__ A) {
    int idx = blockIdx.x * 256 + threadIdx.x;
    if (idx >= MPIX * KCONV) return;
    int c = idx % KCONV;
    int p = idx / KCONV;
    int b = p >> 8, pix = p & 255;
    int oy = pix >> 4, ox = pix & 15;
    int ci = c / 9, kk = c % 9;
    int kh = kk / 3, kw = kk % 3;
    int iy = oy * 2 - 1 + kh;
    int ix = ox * 2 - 1 + kw;
    float v = 0.f;
    if (iy >= 0 && iy < HIMG && ix >= 0 && ix < HIMG)
        v = x[(((long)b * CIN + ci) * HIMG + iy) * HIMG + ix];
    A[idx] = (_Float16)v;
}

// ---------------- main WMMA GEMM -------------------------------------------
// C[M,N] = act(A[M,K] * W[N,K]^T + bias) (+resid), M%64==0, N%128==0, K%32==0.
// Block 128 thr = 4 waves in 2x2; block tile 64x128, wave tile 32x64.
// A/B tiles double-buffered in LDS via global_load_async_to_lds_b128.
__global__ void __launch_bounds__(128) gemm_wmma(
        const _Float16* __restrict__ A, const _Float16* __restrict__ W,
        const float* __restrict__ bias, const float* __restrict__ resid,
        float* __restrict__ outF, _Float16* __restrict__ outH,
        int M, int N, int K, int act) {
    __shared__ _Float16 As[2][64][32];    // 8 KB
    __shared__ _Float16 Bs[2][128][32];   // 16 KB
    const int t = threadIdx.x;
    const int lane = t & 31;
    const int wave = t >> 5;
    const int wm = (wave >> 1) * 32;   // wave row offset in block tile
    const int wn = (wave & 1) * 64;    // wave col offset in block tile
    const long m0 = (long)blockIdx.y * 64;
    const long n0 = (long)blockIdx.x * 128;

    // per-thread async-copy assignment:
    //   A tile 64x32 halves: thread t -> row t>>1, halves [16*(t&1), 16*(t&1)+16)
    //   B tile 128x32 halves: thread t -> row t, all 32 halves
    const int ar = t >> 1;
    const int ac = (t & 1) << 4;
    const _Float16* agp = A + (m0 + ar) * (long)K + ac;
    const _Float16* bgp = W + (n0 + t) * (long)K;

    // prologue: stage k-step 0 into buffer 0
    async_g2l_b128(LDSOFF(&As[0][ar][ac]),     agp);
    async_g2l_b128(LDSOFF(&As[0][ar][ac + 8]), agp + 8);
#pragma unroll
    for (int c = 0; c < 4; ++c)
        async_g2l_b128(LDSOFF(&Bs[0][t][c * 8]), bgp + c * 8);

    v8f acc[2][4] = {};
    const int nsteps = K >> 5;
    for (int s = 0; s < nsteps; ++s) {
        const int buf = s & 1;
        wait_async0();          // my copies into buf are done
        __syncthreads();        // everyone's copies done; prev reads of buf^1 done
        if (s + 1 < nsteps) {   // prefetch next k-step into the other buffer
            const int k1 = (s + 1) << 5;
            async_g2l_b128(LDSOFF(&As[buf ^ 1][ar][ac]),     agp + k1);
            async_g2l_b128(LDSOFF(&As[buf ^ 1][ar][ac + 8]), agp + k1 + 8);
#pragma unroll
            for (int c = 0; c < 4; ++c)
                async_g2l_b128(LDSOFF(&Bs[buf ^ 1][t][c * 8]), bgp + k1 + c * 8);
        }
        v16h af[2], bf[4];
#pragma unroll
        for (int i = 0; i < 2; ++i)
            af[i] = load_a_frag(&As[buf][0][0], 32, wm + i * 16, 0, lane);
#pragma unroll
        for (int j = 0; j < 4; ++j)
            bf[j] = load_b_frag(&Bs[buf][0][0], 32, wn + j * 16, 0, lane);
#pragma unroll
        for (int i = 0; i < 2; ++i)
#pragma unroll
            for (int j = 0; j < 4; ++j)
                acc[i][j] = __builtin_amdgcn_wmma_f32_16x16x32_f16(
                    false, af[i], false, bf[j], (short)0, acc[i][j], false, false);
    }

    // store: lane L holds col (L&15), rows r + 8*(L>>4)
#pragma unroll
    for (int i = 0; i < 2; ++i) {
        const long rbase = m0 + wm + i * 16 + ((lane >> 4) << 3);
#pragma unroll
        for (int j = 0; j < 4; ++j) {
            const int col = (int)n0 + wn + j * 16 + (lane & 15);
            float bv = bias ? bias[col] : 0.f;
#pragma unroll
            for (int r = 0; r < 8; ++r) {
                long row = rbase + r;
                float v = acc[i][j][r] + bv;
                if (act == 1) v = gelu_exact(v);
                if (resid) v += resid[row * N + col];
                if (outF) outF[row * N + col] = v;
                if (outH) outH[row * N + col] = (_Float16)v;
            }
        }
    }
}

// ---------------- patch-embed LN + extra token + geo bias --------------------
__global__ void embed_ln(const float* __restrict__ convout, const float* __restrict__ pe_g,
                         const float* __restrict__ pe_b, const float* __restrict__ etok,
                         const float* __restrict__ geo, float* __restrict__ y) {
    int row = blockIdx.x;
    int b = row / NTOK, n = row % NTOK;
    float* out = y + (long)row * DM;
    const float* g = geo + (long)n * DM;
    if (n == 0) {
        for (int d = threadIdx.x; d < DM; d += 256) out[d] = etok[d] + g[d];
        return;
    }
    const float* src = convout + (long)(b * 256 + (n - 1)) * DM;
    float s = 0.f;
    for (int d = threadIdx.x; d < DM; d += 256) s += src[d];
    float mean = block_sum(s) * (1.0f / DM);
    float vs = 0.f;
    for (int d = threadIdx.x; d < DM; d += 256) { float t = src[d] - mean; vs += t * t; }
    float rstd = rsqrtf(block_sum(vs) * (1.0f / DM) + 1e-5f);
    for (int d = threadIdx.x; d < DM; d += 256)
        out[d] = (src[d] - mean) * rstd * pe_g[d] + pe_b[d] + g[d];
}

// LayerNorm rows of f32 [rows][768] -> f16
__global__ void ln_rows_f16(const float* __restrict__ in, const float* __restrict__ gg,
                            const float* __restrict__ bb, _Float16* __restrict__ out) {
    int row = blockIdx.x;
    const float* src = in + (long)row * DM;
    _Float16* dst = out + (long)row * DM;
    float s = 0.f;
    for (int d = threadIdx.x; d < DM; d += 256) s += src[d];
    float mean = block_sum(s) * (1.0f / DM);
    float vs = 0.f;
    for (int d = threadIdx.x; d < DM; d += 256) { float t = src[d] - mean; vs += t * t; }
    float rstd = rsqrtf(block_sum(vs) * (1.0f / DM) + 1e-5f);
    for (int d = threadIdx.x; d < DM; d += 256)
        dst[d] = (_Float16)((src[d] - mean) * rstd * gg[d] + bb[d]);
}

// ---------------- qkv repack: f32 [B*257][2304] -> padded f16 heads ----------
__global__ void repack_qkv(const float* __restrict__ qkvF, _Float16* __restrict__ qh,
                           _Float16* __restrict__ kh, _Float16* __restrict__ vT) {
    int idx = blockIdx.x * 256 + threadIdx.x;
    if (idx >= BQ * HEADS * NP * HD) return;
    int d = idx & 63;
    int n = (idx >> 6) % NP;
    int bh = idx / (NP * HD);
    int b = bh / HEADS, h = bh % HEADS;
    float qv = 0.f, kv = 0.f, vv = 0.f;
    if (n < NTOK) {
        const float* rowp = qkvF + (long)(b * NTOK + n) * (3 * DM);
        qv = rowp[0 * DM + h * HD + d] * 0.125f;   // HD^-0.5
        kv = rowp[1 * DM + h * HD + d];
        vv = rowp[2 * DM + h * HD + d];
    }
    qh[idx] = (_Float16)qv;
    kh[idx] = (_Float16)kv;
    vT[((long)bh * HD + d) * NP + n] = (_Float16)vv;
}

// ---------------- attention scores: S = q k^T + relpos bias ------------------
__global__ void attn_scores(const _Float16* __restrict__ qh, const _Float16* __restrict__ kh,
                            const float* __restrict__ rpb, float* __restrict__ S) {
    int bh = blockIdx.y;
    int h = bh % HEADS;
    int tn = blockIdx.x / 17;    // query tile
    int tm = blockIdx.x % 17;    // key tile
    int lane = threadIdx.x & 31;
    const _Float16* q = qh + (long)bh * NP * HD;
    const _Float16* k = kh + (long)bh * NP * HD;
    v8f acc = {};
#pragma unroll
    for (int k0 = 0; k0 < HD; k0 += 32) {
        v16h a = load_a_frag(q, HD, tn * 16, k0, lane);
        v16h b = load_b_frag(k, HD, tm * 16, k0, lane);
        acc = __builtin_amdgcn_wmma_f32_16x16x32_f16(false, a, false, b,
                                                     (short)0, acc, false, false);
    }
    int m = tm * 16 + (lane & 15);
    int nb = tn * 16 + ((lane >> 4) << 3);
#pragma unroll
    for (int r = 0; r < 8; ++r) {
        int n = nb + r;
        int idx = 0;
        if (n >= 1 && n < NTOK && m >= 1 && m < NTOK) {
            int p = n - 1, qq = m - 1;
            int dy = (p >> 4) - (qq >> 4) + (GSZ - 1);
            int dx = (p & 15) - (qq & 15) + (GSZ - 1);
            idx = dy * (2 * GSZ - 1) + dx;
        }
        S[((long)bh * NP + n) * NP + m] = acc[r] + rpb[idx * HEADS + h];
    }
}

// ---------------- softmax rows (one wave per row) ----------------------------
__global__ void softmax_rows(const float* __restrict__ S, _Float16* __restrict__ P) {
    int row = blockIdx.x;
    int bh = row / 272, n = row % 272;
    long base = ((long)bh * NP + n) * NP;
    int lane = threadIdx.x & 31;
    if (n >= NTOK) {
        for (int m = lane; m < NP; m += 32) P[base + m] = (_Float16)0.0f;
        return;
    }
    const float* s = S + base;
    float vals[9];
    float mx = -3.0e38f;
#pragma unroll
    for (int i = 0; i < 9; ++i) {
        int m = lane + i * 32;
        vals[i] = (m < NTOK) ? s[m] : -3.0e38f;
        mx = fmaxf(mx, vals[i]);
    }
#pragma unroll
    for (int off = 16; off > 0; off >>= 1) mx = fmaxf(mx, __shfl_xor(mx, off, 32));
    float sum = 0.f;
#pragma unroll
    for (int i = 0; i < 9; ++i) {
        int m = lane + i * 32;
        float e = (m < NTOK) ? __expf(vals[i] - mx) : 0.f;
        vals[i] = e;
        sum += e;
    }
#pragma unroll
    for (int off = 16; off > 0; off >>= 1) sum += __shfl_xor(sum, off, 32);
    float inv = 1.0f / sum;
#pragma unroll
    for (int i = 0; i < 9; ++i) {
        int m = lane + i * 32;
        if (m < NP) P[base + m] = (_Float16)(vals[i] * inv);
    }
}

// ---------------- O = P * V  -> o16 [B*257][768] -----------------------------
__global__ void attn_pv(const _Float16* __restrict__ P, const _Float16* __restrict__ vT,
                        _Float16* __restrict__ o16) {
    int bh = blockIdx.y;
    int b = bh / HEADS, h = bh % HEADS;
    int tr = blockIdx.x >> 2;
    int tc = blockIdx.x & 3;
    int lane = threadIdx.x & 31;
    const _Float16* Pp = P + (long)bh * NP * NP;
    const _Float16* Vp = vT + (long)bh * HD * NP;
    v8f acc = {};
    for (int k0 = 0; k0 < NP; k0 += 32) {
        v16h a = load_a_frag(Pp, NP, tr * 16, k0, lane);
        v16h bb = load_b_frag(Vp, NP, tc * 16, k0, lane);
        acc = __builtin_amdgcn_wmma_f32_16x16x32_f16(false, a, false, bb,
                                                     (short)0, acc, false, false);
    }
    int d = tc * 16 + (lane & 15);
    int nb = tr * 16 + ((lane >> 4) << 3);
#pragma unroll
    for (int r = 0; r < 8; ++r) {
        int n = nb + r;
        if (n < NTOK)
            o16[((long)(b * NTOK + n)) * DM + h * HD + d] = (_Float16)acc[r];
    }
}

// ---------------------------------------------------------------------------
extern "C" void kernel_launch(void* const* d_in, const int* in_sizes, int n_in,
                              void* d_out, int out_size, void* d_ws, size_t ws_size,
                              hipStream_t stream) {
    const float* x      = (const float*)d_in[0];
    const float* geo    = (const float*)d_in[3];
    const float* etok   = (const float*)d_in[4];
    const float* conv_w = (const float*)d_in[5];
    const float* conv_b = (const float*)d_in[6];
    const float* pe_g   = (const float*)d_in[7];
    const float* pe_b   = (const float*)d_in[8];
    const float* n1_g   = (const float*)d_in[9];
    const float* n1_b   = (const float*)d_in[10];
    const float* qkv_w  = (const float*)d_in[11];
    const float* rpb    = (const float*)d_in[12];
    const float* proj_w = (const float*)d_in[13];
    const float* proj_b = (const float*)d_in[14];
    const float* n2_g   = (const float*)d_in[15];
    const float* n2_b   = (const float*)d_in[16];
    const float* fc1_w  = (const float*)d_in[17];
    const float* fc1_b  = (const float*)d_in[18];
    const float* fc2_w  = (const float*)d_in[19];
    const float* fc2_b  = (const float*)d_in[20];
    float* out = (float*)d_out;

    char* ws = (char*)d_ws;
    size_t cur = 0;
    auto alloc = [&](size_t bytes) -> void* {
        void* p = ws + cur;
        cur += (bytes + 255) & ~(size_t)255;
        return p;
    };

    _Float16* Wc16   = (_Float16*)alloc((size_t)DM * KCONV * 2);
    _Float16* Wqkv16 = (_Float16*)alloc((size_t)3 * DM * DM * 2);
    _Float16* Wprj16 = (_Float16*)alloc((size_t)DM * DM * 2);
    _Float16* Wfc116 = (_Float16*)alloc((size_t)MLPH * DM * 2);
    _Float16* Wfc216 = (_Float16*)alloc((size_t)DM * MLPH * 2);
    _Float16* A16    = (_Float16*)alloc((size_t)MPIX * KCONV * 2);
    _Float16* h16    = (_Float16*)alloc((size_t)MTOK * DM * 2);
    _Float16* qh16   = (_Float16*)alloc((size_t)BQ * HEADS * NP * HD * 2);
    _Float16* kh16   = (_Float16*)alloc((size_t)BQ * HEADS * NP * HD * 2);
    _Float16* vT16   = (_Float16*)alloc((size_t)BQ * HEADS * HD * NP * 2);
    _Float16* P16    = (_Float16*)alloc((size_t)BQ * HEADS * NP * NP * 2);
    _Float16* o16    = (_Float16*)alloc((size_t)MTOK * DM * 2);
    _Float16* h216   = (_Float16*)alloc((size_t)MTOK * DM * 2);
    _Float16* mid16  = (_Float16*)alloc((size_t)MTOK * MLPH * 2);
    float*    convF  = (float*)alloc((size_t)MPIX * DM * 4);
    float*    yF     = (float*)alloc((size_t)MTOK * DM * 4);
    float*    qkvF   = (float*)alloc((size_t)MTOK * 3 * DM * 4);
    float*    SF     = (float*)alloc((size_t)BQ * HEADS * NP * NP * 4);

    auto cvt = [&](const float* s, _Float16* d, int n) {
        cvt_f32_f16<<<(n + 255) / 256, 256, 0, stream>>>(s, d, n);
    };
    cvt(conv_w, Wc16,   DM * KCONV);
    cvt(qkv_w,  Wqkv16, 3 * DM * DM);
    cvt(proj_w, Wprj16, DM * DM);
    cvt(fc1_w,  Wfc116, MLPH * DM);
    cvt(fc2_w,  Wfc216, DM * MLPH);

    // ---- patch embed: im2col + WMMA GEMM + LN + tokens ----
    {
        long n = (long)MPIX * KCONV;
        im2col_f16<<<(int)((n + 255) / 256), 256, 0, stream>>>(x, A16);
    }
    gemm_wmma<<<dim3(DM / 128, MPIX / 64), 128, 0, stream>>>(
        A16, Wc16, conv_b, nullptr, convF, nullptr, MPIX, DM, KCONV, 0);
    embed_ln<<<MTOK, 256, 0, stream>>>(convF, pe_g, pe_b, etok, geo, yF);

    // ---- attention ----
    ln_rows_f16<<<MTOK, 256, 0, stream>>>(yF, n1_g, n1_b, h16);
    gemm_wmma<<<dim3(3 * DM / 128, MTOK / 64), 128, 0, stream>>>(
        h16, Wqkv16, nullptr, nullptr, qkvF, nullptr, MTOK, 3 * DM, DM, 0);
    {
        int n = BQ * HEADS * NP * HD;
        repack_qkv<<<(n + 255) / 256, 256, 0, stream>>>(qkvF, qh16, kh16, vT16);
    }
    attn_scores<<<dim3(17 * 17, BQ * HEADS), 32, 0, stream>>>(qh16, kh16, rpb, SF);
    softmax_rows<<<BQ * HEADS * 272, 32, 0, stream>>>(SF, P16);
    attn_pv<<<dim3(17 * 4, BQ * HEADS), 32, 0, stream>>>(P16, vT16, o16);
    gemm_wmma<<<dim3(DM / 128, MTOK / 64), 128, 0, stream>>>(
        o16, Wprj16, proj_b, yF, yF, nullptr, MTOK, DM, DM, 0);

    // ---- MLP ----
    ln_rows_f16<<<MTOK, 256, 0, stream>>>(yF, n2_g, n2_b, h216);
    gemm_wmma<<<dim3(MLPH / 128, MTOK / 64), 128, 0, stream>>>(
        h216, Wfc116, fc1_b, nullptr, nullptr, mid16, MTOK, MLPH, DM, 1);
    gemm_wmma<<<dim3(DM / 128, MTOK / 64), 128, 0, stream>>>(
        mid16, Wfc216, fc2_b, yF, out, nullptr, MTOK, DM, MLPH, 0);
}
